// VideoMusicTransformer_V1_68728066670836
// MI455X (gfx1250) — matrix operand order repaired and checked by reading, not compiled, not run on hardware
//
#include <hip/hip_runtime.h>
#include <hip/hip_bf16.h>

// ---------------------------------------------------------------------------
// VideoMusicTransformer forward on MI455X (gfx1250, wave32).
// All GEMMs go through a bf16 WMMA tiled kernel (v_wmma_f32_16x16x32_bf16):
//   C[M,N] = A[M,K] * B + bias,  B either [N,K] (K-contig, "nk") or [K,N]
//   ("kn"), with optional 2-level batch strides (batch b, head h) so the same
//   kernel serves token projections, per-(b,h) QK^T and P*V.
// Tile staging: interior tiles use float4 global loads -> packed bf16 ->
// vector LDS stores; edge tiles use a branch-free clamped path.
// ---------------------------------------------------------------------------

typedef long long ll;
typedef __attribute__((ext_vector_type(16))) __bf16 v16bf;
typedef __attribute__((ext_vector_type(8)))  __bf16 v8bf;
typedef __attribute__((ext_vector_type(4)))  __bf16 v4bf;
typedef __attribute__((ext_vector_type(8)))  float  v8f;

typedef struct { float x, y, z, w; } f4;

#define TM 64
#define TN 64
#define TK 32
#define KP 48   // LDS row stride (elements): 96B rows, keeps 8B/16B/32B alignment

__device__ __forceinline__ int iminz(int a, int b) { return a < b ? a : b; }

// ------------------------------ GEMM (WMMA) --------------------------------
__global__ __launch_bounds__(128)
void gemm_wmma_kernel(const float* __restrict__ A, const float* __restrict__ Bw,
                      const float* __restrict__ bias, float* __restrict__ C,
                      int M, int N, int K, int lda, int ldb, int ldc,
                      int nh, ll sAb, ll sAh, ll sBb, ll sBh, ll sCb, ll sCh,
                      int b_is_kn)
{
    __shared__ __bf16 As[TM][KP];
    __shared__ __bf16 Bs[TN][KP];

    const int tid  = threadIdx.x;
    const int lane = tid & 31;
    const int wave = tid >> 5;

    const int z  = blockIdx.z;
    const int zb = z / nh;
    const int zh = z % nh;
    A  += (ll)zb * sAb + (ll)zh * sAh;
    Bw += (ll)zb * sBb + (ll)zh * sBh;
    C  += (ll)zb * sCb + (ll)zh * sCh;

    const int m0 = blockIdx.y * TM;
    const int n0 = blockIdx.x * TN;
    const int wm = (wave >> 1) * 32;   // 2x2 wave grid of 32x32 tiles
    const int wn = (wave & 1) * 32;

    const bool mfull = (m0 + TM <= M);
    const bool nfull = (n0 + TN <= N);
    const bool afast = mfull && ((lda & 3) == 0);
    const bool bfast = nfull && ((ldb & 3) == 0);

    v8f acc[2][2];
    const v8f vzero = {0.f,0.f,0.f,0.f,0.f,0.f,0.f,0.f};
    acc[0][0] = vzero; acc[0][1] = vzero; acc[1][0] = vzero; acc[1][1] = vzero;

    const int lr = lane & 15;
    const int lh = lane >> 4;

    for (int k0 = 0; k0 < K; k0 += TK) {
        const bool kfull = (k0 + TK <= K);

        // ---------------- stage A tile: As[r][k] (row-major) ----------------
        if (afast && kfull) {
            #pragma unroll
            for (int it = 0; it < 4; ++it) {
                int idx = tid + it * 128;          // 512 float4 chunks
                int r  = idx >> 3;                 // 8 chunks per 32-wide row
                int c4 = (idx & 7) << 2;
                const float* src = A + (ll)(m0 + r) * lda + k0 + c4;
                const f4 v = *(const f4*)src;
                v4bf p; p[0] = (__bf16)v.x; p[1] = (__bf16)v.y;
                        p[2] = (__bf16)v.z; p[3] = (__bf16)v.w;
                *(v4bf*)&As[r][c4] = p;
                if (k0 + TK < K) __builtin_prefetch(src + TK, 0, 3);
            }
        } else {
            for (int idx = tid; idx < TM * TK; idx += 128) {
                int r = idx >> 5, c = idx & 31;
                int gm = m0 + r, gk = k0 + c;
                bool ok = (gm < M) & (gk < K);
                float v = A[(ll)iminz(gm, M - 1) * lda + iminz(gk, K - 1)];
                As[r][c] = (__bf16)(ok ? v : 0.f);
            }
        }

        // ---------------- stage B tile as Bs[n][k] --------------------------
        if (b_is_kn) {
            if (bfast && kfull) {
                #pragma unroll
                for (int it = 0; it < 4; ++it) {
                    int idx = tid + it * 128;      // 512 float4 chunks
                    int kk = idx >> 4;             // 16 chunks per 64-wide row
                    int n4 = (idx & 15) << 2;
                    const float* src = Bw + (ll)(k0 + kk) * ldb + n0 + n4;
                    const f4 v = *(const f4*)src;  // coalesced over n
                    Bs[n4 + 0][kk] = (__bf16)v.x;  // transpose into LDS
                    Bs[n4 + 1][kk] = (__bf16)v.y;
                    Bs[n4 + 2][kk] = (__bf16)v.z;
                    Bs[n4 + 3][kk] = (__bf16)v.w;
                    if (k0 + TK < K) __builtin_prefetch(src + (ll)TK * ldb, 0, 3);
                }
            } else {
                for (int idx = tid; idx < TN * TK; idx += 128) {
                    int kk = idx >> 6, nn = idx & 63;
                    int gk = k0 + kk, gn = n0 + nn;
                    bool ok = (gk < K) & (gn < N);
                    float v = Bw[(ll)iminz(gk, K - 1) * ldb + iminz(gn, N - 1)];
                    Bs[nn][kk] = (__bf16)(ok ? v : 0.f);
                }
            }
        } else {
            if (bfast && kfull) {
                #pragma unroll
                for (int it = 0; it < 4; ++it) {
                    int idx = tid + it * 128;      // 512 float4 chunks
                    int nn = idx >> 3;             // 8 chunks per 32-wide row
                    int c4 = (idx & 7) << 2;
                    const float* src = Bw + (ll)(n0 + nn) * ldb + k0 + c4;
                    const f4 v = *(const f4*)src;
                    v4bf p; p[0] = (__bf16)v.x; p[1] = (__bf16)v.y;
                            p[2] = (__bf16)v.z; p[3] = (__bf16)v.w;
                    *(v4bf*)&Bs[nn][c4] = p;
                    if (k0 + TK < K) __builtin_prefetch(src + TK, 0, 3);
                }
            } else {
                for (int idx = tid; idx < TN * TK; idx += 128) {
                    int nn = idx >> 5, kk = idx & 31;
                    int gn = n0 + nn, gk = k0 + kk;
                    bool ok = (gn < N) & (gk < K);
                    float v = Bw[(ll)iminz(gn, N - 1) * ldb + iminz(gk, K - 1)];
                    Bs[nn][kk] = (__bf16)(ok ? v : 0.f);
                }
            }
        }
        __syncthreads();

        // --- build fragments per documented 16-bit WMMA layouts ---
        // A 16x32: lanes 0-15 row M=lane, K {0..7,16..23}; lanes 16-31 same M,
        //          K {8..15,24..31}.
        v16bf afrag[2], bfrag[2];
        #pragma unroll
        for (int f = 0; f < 2; ++f) {
            const v8bf a0 = *(const v8bf*)&As[wm + f * 16 + lr][lh * 8];
            const v8bf a1 = *(const v8bf*)&As[wm + f * 16 + lr][16 + lh * 8];
            afrag[f] = __builtin_shufflevector(a0, a1,
                0,1,2,3,4,5,6,7,8,9,10,11,12,13,14,15);
            // B 32x16: lanes 0-15 col N=lane, K {0..7,16..23}; lanes 16-31 K {8..15,24..31}.
            const v8bf b0 = *(const v8bf*)&Bs[wn + f * 16 + lr][lh * 8];
            const v8bf b1 = *(const v8bf*)&Bs[wn + f * 16 + lr][16 + lh * 8];
            bfrag[f] = __builtin_shufflevector(b0, b1,
                0,1,2,3,4,5,6,7,8,9,10,11,12,13,14,15);
        }
        #pragma unroll
        for (int fm = 0; fm < 2; ++fm)
            #pragma unroll
            for (int fn = 0; fn < 2; ++fn)
                acc[fm][fn] = __builtin_amdgcn_wmma_f32_16x16x32_bf16(
                    false, afrag[fm], false, bfrag[fn],
                    (short)0, acc[fm][fn], false, false);
        __syncthreads();
    }

    // --- store: VGPR r -> row r (lanes 0-15) / r+8 (lanes 16-31), col=lane&15
    const int col   = lane & 15;
    const int rbase = (lane >> 4) * 8;
    #pragma unroll
    for (int fm = 0; fm < 2; ++fm)
        #pragma unroll
        for (int fn = 0; fn < 2; ++fn) {
            int gn = n0 + wn + fn * 16 + col;
            float bv = (bias && gn < N) ? bias[gn] : 0.f;
            #pragma unroll
            for (int r = 0; r < 8; ++r) {
                int gm = m0 + wm + fm * 16 + rbase + r;
                if (gm < M && gn < N)
                    C[(ll)gm * ldc + gn] = acc[fm][fn][r] + bv;
            }
        }
}

// --------------------------- elementwise kernels ---------------------------
__global__ void pack_vfc_kernel(const float* __restrict__ fsem,
                                const float* __restrict__ fscene,
                                const float* __restrict__ fmotion,
                                const float* __restrict__ femotion,
                                float* __restrict__ out, ll n)
{
    ll i = (ll)blockIdx.x * 256 + threadIdx.x;
    if (i >= n) return;
    ll t = i / 776; int j = (int)(i % 776);
    float v;
    if (j < 768)       v = fsem[t * 768 + j];
    else if (j == 768) v = fscene[t];
    else if (j == 769) v = fmotion[t];
    else               v = femotion[t * 6 + (j - 770)];
    out[i] = v;
}

__global__ void pack_chord_kernel(const int* __restrict__ x_root,
                                  const int* __restrict__ x_attr,
                                  const float* __restrict__ fkey,
                                  const float* __restrict__ emb_root,
                                  const float* __restrict__ emb_attr,
                                  float* __restrict__ out, int SCd, ll n)
{
    ll i = (ll)blockIdx.x * 256 + threadIdx.x;
    if (i >= n) return;
    ll t = i / 513; int j = (int)(i % 513);
    float v;
    if (j < 512) v = emb_root[(ll)x_root[t] * 512 + j] + emb_attr[(ll)x_attr[t] * 512 + j];
    else         v = fkey[t / SCd];
    out[i] = v;
}

__global__ void add_pos_kernel(float* __restrict__ h, const float* __restrict__ pos,
                               int S, int Dd, ll n)
{
    ll i = (ll)blockIdx.x * 256 + threadIdx.x;
    if (i >= n) return;
    int d = (int)(i % Dd);
    int s = (int)((i / Dd) % S);
    h[i] += pos[(ll)s * Dd + d];
}

__global__ void silu_mul_kernel(float* __restrict__ h1, const float* __restrict__ h2, ll n)
{
    ll i = (ll)blockIdx.x * 256 + threadIdx.x;
    if (i >= n) return;
    float x = h1[i];
    float s = x / (1.f + __expf(-x));
    h1[i] = s * h2[i];
}

__global__ void combine_kernel(float* __restrict__ acc, const float* __restrict__ eo,
                               const float* __restrict__ cw, int e, int Ee, int Dd,
                               ll n, int overwrite)
{
    ll i = (ll)blockIdx.x * 256 + threadIdx.x;
    if (i >= n) return;
    ll t = i / Dd;
    float v = cw[t * Ee + e] * eo[i];
    acc[i] = overwrite ? v : (acc[i] + v);
}

// residual + layernorm: out = LN(x + a) * g + b ; a may be null (plain LN)
__global__ __launch_bounds__(128)
void resln_kernel(const float* __restrict__ x, const float* __restrict__ a,
                  const float* __restrict__ g, const float* __restrict__ bta,
                  float* __restrict__ out, int Dd)
{
    ll t = blockIdx.x;
    const float* px = x + t * (ll)Dd;
    const float* pa = a ? a + t * (ll)Dd : nullptr;
    float* po = out + t * (ll)Dd;
    __shared__ float red[128];
    int tid = threadIdx.x;

    float s = 0.f;
    for (int i = tid; i < Dd; i += 128) s += px[i] + (pa ? pa[i] : 0.f);
    red[tid] = s; __syncthreads();
    for (int o = 64; o; o >>= 1) { if (tid < o) red[tid] += red[tid + o]; __syncthreads(); }
    float mean = red[0] / Dd; __syncthreads();

    float vs = 0.f;
    for (int i = tid; i < Dd; i += 128) {
        float v = px[i] + (pa ? pa[i] : 0.f) - mean; vs += v * v;
    }
    red[tid] = vs; __syncthreads();
    for (int o = 64; o; o >>= 1) { if (tid < o) red[tid] += red[tid + o]; __syncthreads(); }
    float rstd = rsqrtf(red[0] / Dd + 1e-5f); __syncthreads();

    for (int i = tid; i < Dd; i += 128) {
        float v = px[i] + (pa ? pa[i] : 0.f);
        po[i] = (v - mean) * rstd * g[i] + bta[i];
    }
}

// scaled softmax over score rows; causal masks k > q (writes exact zeros)
__global__ __launch_bounds__(128)
void softmax_kernel(float* __restrict__ sc, int Sq, int Sk, float scale, int causal)
{
    int q = blockIdx.x;
    ll row = (ll)blockIdx.y * Sq + q;
    float* p = sc + row * (ll)Sk;
    int limit = causal ? (q + 1) : Sk;
    __shared__ float red[128];
    int tid = threadIdx.x;

    float mx = -3.0e38f;
    for (int i = tid; i < limit; i += 128) mx = fmaxf(mx, p[i] * scale);
    red[tid] = mx; __syncthreads();
    for (int o = 64; o; o >>= 1) { if (tid < o) red[tid] = fmaxf(red[tid], red[tid + o]); __syncthreads(); }
    mx = red[0]; __syncthreads();

    float sum = 0.f;
    for (int i = tid; i < limit; i += 128) {
        float e = __expf(p[i] * scale - mx);
        p[i] = e; sum += e;
    }
    red[tid] = sum; __syncthreads();
    for (int o = 64; o; o >>= 1) { if (tid < o) red[tid] += red[tid + o]; __syncthreads(); }
    float inv = 1.f / red[0]; __syncthreads();

    for (int i = tid; i < Sk; i += 128) p[i] = (i < limit) ? p[i] * inv : 0.f;
}

// top-2 gate: logits = h @ gw^T + gb, softmax, top-2 renormalized -> cw[T,E]
__global__ __launch_bounds__(128)
void gate_kernel(const float* __restrict__ h, const float* __restrict__ gw,
                 const float* __restrict__ gb, float* __restrict__ cw,
                 int Dd, int Ee)
{
    ll t = blockIdx.x;
    const float* px = h + t * (ll)Dd;
    int tid = threadIdx.x;
    float part[8];
    for (int e = 0; e < 8; ++e) part[e] = 0.f;
    for (int i = tid; i < Dd; i += 128) {
        float xv = px[i];
        for (int e = 0; e < Ee; ++e) part[e] += xv * gw[(ll)e * Dd + i];
    }
    __shared__ float red[8 * 128];
    for (int e = 0; e < Ee; ++e) red[e * 128 + tid] = part[e];
    __syncthreads();
    for (int o = 64; o; o >>= 1) {
        if (tid < o)
            for (int e = 0; e < Ee; ++e) red[e * 128 + tid] += red[e * 128 + tid + o];
        __syncthreads();
    }
    if (tid == 0) {
        float lg[8]; float mx = -3.0e38f;
        for (int e = 0; e < Ee; ++e) { lg[e] = red[e * 128] + gb[e]; mx = fmaxf(mx, lg[e]); }
        float sum = 0.f;
        for (int e = 0; e < Ee; ++e) { lg[e] = __expf(lg[e] - mx); sum += lg[e]; }
        for (int e = 0; e < Ee; ++e) lg[e] /= sum;
        int i0 = 0;
        for (int e = 1; e < Ee; ++e) if (lg[e] > lg[i0]) i0 = e;
        int i1 = -1;
        for (int e = 0; e < Ee; ++e) if (e != i0 && (i1 < 0 || lg[e] > lg[i1])) i1 = e;
        float w0 = lg[i0], w1 = lg[i1], wsum = w0 + w1;
        w0 /= wsum; w1 /= wsum;
        for (int e = 0; e < Ee; ++e)
            cw[t * Ee + e] = (e == i0) ? w0 : ((e == i1) ? w1 : 0.f);
    }
}

// ------------------------------- host side ---------------------------------
extern "C" void kernel_launch(void* const* d_in, const int* in_sizes, int n_in,
                              void* d_out, int out_size, void* d_ws, size_t ws_size,
                              hipStream_t stream)
{
    (void)in_sizes; (void)n_in; (void)out_size; (void)ws_size;

    constexpr int Bq = 32, SCd = 300, SVd = 300, Dd = 512, Hh = 8, Ln = 6,
                  Ee = 6, Ff = 1024, HD = 64, CHORD = 159;
    constexpr int T = Bq * SCd;          // 9600 tokens (enc and dec equal)
    const float scale = 0.125f;          // 1/sqrt(64)

    // ---- inputs (setup_inputs dict order) ----
    const int*   x_root   = (const int*)  d_in[1];
    const int*   x_attr   = (const int*)  d_in[2];
    const float* fsem     = (const float*)d_in[3];
    const float* fkey     = (const float*)d_in[4];
    const float* fscene   = (const float*)d_in[5];
    const float* fmotion  = (const float*)d_in[6];
    const float* femotion = (const float*)d_in[7];
    const float* emb_root = (const float*)d_in[8];
    const float* emb_attr = (const float*)d_in[9];
    const float* w_chord  = (const float*)d_in[10];
    const float* b_chord  = (const float*)d_in[11];
    const float* w_vis    = (const float*)d_in[12];
    const float* b_vis    = (const float*)d_in[13];
    const float* pos_c    = (const float*)d_in[14];
    const float* pos_v    = (const float*)d_in[15];
    const float* enc_qkv_w = (const float*)d_in[16];
    const float* enc_qkv_b = (const float*)d_in[17];
    const float* enc_ow    = (const float*)d_in[18];
    const float* enc_ob    = (const float*)d_in[19];
    const float* dec_sqkv_w = (const float*)d_in[20];
    const float* dec_sqkv_b = (const float*)d_in[21];
    const float* dec_sow    = (const float*)d_in[22];
    const float* dec_sob    = (const float*)d_in[23];
    const float* dec_cqkv_w = (const float*)d_in[24];
    const float* dec_cqkv_b = (const float*)d_in[25];
    const float* dec_cow    = (const float*)d_in[26];
    const float* dec_cob    = (const float*)d_in[27];
    const float* enc_gw = (const float*)d_in[28];
    const float* enc_gb = (const float*)d_in[29];
    const float* enc_w1 = (const float*)d_in[30];
    const float* enc_b1 = (const float*)d_in[31];
    const float* enc_w2 = (const float*)d_in[32];
    const float* enc_b2 = (const float*)d_in[33];
    const float* enc_w3 = (const float*)d_in[34];
    const float* enc_b3 = (const float*)d_in[35];
    const float* dec_gw = (const float*)d_in[36];
    const float* dec_gb = (const float*)d_in[37];
    const float* dec_w1 = (const float*)d_in[38];
    const float* dec_b1 = (const float*)d_in[39];
    const float* dec_w2 = (const float*)d_in[40];
    const float* dec_b2 = (const float*)d_in[41];
    const float* dec_w3 = (const float*)d_in[42];
    const float* dec_b3 = (const float*)d_in[43];
    const float* enc_ln1_g = (const float*)d_in[44];
    const float* enc_ln1_b = (const float*)d_in[45];
    const float* enc_ln2_g = (const float*)d_in[46];
    const float* enc_ln2_b = (const float*)d_in[47];
    const float* dec_ln1_g = (const float*)d_in[48];
    const float* dec_ln1_b = (const float*)d_in[49];
    const float* dec_ln2_g = (const float*)d_in[50];
    const float* dec_ln2_b = (const float*)d_in[51];
    const float* dec_ln3_g = (const float*)d_in[52];
    const float* dec_ln3_b = (const float*)d_in[53];
    const float* enc_fg = (const float*)d_in[54];
    const float* enc_fb = (const float*)d_in[55];
    const float* dec_fg = (const float*)d_in[56];
    const float* dec_fb = (const float*)d_in[57];
    const float* wout = (const float*)d_in[58];
    const float* bout = (const float*)d_in[59];
    float* out = (float*)d_out;

    // ---- workspace carve-out (floats) ----
    float* ws = (float*)d_ws;
    size_t off = 0;
    auto carve = [&](size_t n) { float* p = ws + off; off += n; return p; };
    float* hEnc = carve((size_t)T * Dd);
    float* hDec = carve((size_t)T * Dd);
    float* mem  = carve((size_t)T * Dd);
    float* qb   = carve((size_t)T * Dd);
    float* kb   = carve((size_t)T * Dd);
    float* vb   = carve((size_t)T * Dd);
    float* t1   = carve((size_t)T * Dd);                 // attn ctx / expert out
    float* t2   = carve((size_t)T * Dd);                 // attn out / moe accum
    float* sc   = carve((size_t)Bq * Hh * SCd * SCd);    // scores [B,H,Sq,Sk]
    float* h1   = carve((size_t)T * Ff);                 // also vfc staging
    float* h2   = carve((size_t)T * Ff);                 // also chord staging
    float* cw   = carve((size_t)T * Ee);

    auto blocks = [](ll n) { return (unsigned)((n + 255) / 256); };

    auto gemm = [&](const float* A, const float* Bw, const float* bias, float* C,
                    int M, int N, int K, int lda, int ldb, int ldc, int bkn,
                    int nz = 1, int nh = 1,
                    ll sAb = 0, ll sAh = 0, ll sBb = 0, ll sBh = 0,
                    ll sCb = 0, ll sCh = 0) {
        dim3 grid((N + TN - 1) / TN, (M + TM - 1) / TM, nz);
        gemm_wmma_kernel<<<grid, 128, 0, stream>>>(
            A, Bw, bias, C, M, N, K, lda, ldb, ldc,
            nh, sAb, sAh, sBb, sBh, sCb, sCh, bkn);
    };

    // full MHA: q from hq, k/v from hkv; writes projected output to outbuf
    auto attention = [&](const float* hq, const float* hkv,
                         const float* qkvw, const float* qkvb,
                         const float* ow, const float* ob,
                         int causal, float* outbuf) {
        gemm(hq,  qkvw,               qkvb,          qb, T, Dd, Dd, Dd, Dd, Dd, 0);
        gemm(hkv, qkvw + (ll)Dd * Dd,     qkvb + Dd,     kb, T, Dd, Dd, Dd, Dd, Dd, 0);
        gemm(hkv, qkvw + (ll)2 * Dd * Dd, qkvb + 2 * Dd, vb, T, Dd, Dd, Dd, Dd, Dd, 0);
        // scores[b,h] = Q_h @ K_h^T   (K=64)
        gemm(qb, kb, nullptr, sc, SCd, SCd, HD, Dd, Dd, SCd, /*bkn=*/0,
             Bq * Hh, Hh,
             (ll)SCd * Dd, HD, (ll)SCd * Dd, HD,
             (ll)Hh * SCd * SCd, (ll)SCd * SCd);
        softmax_kernel<<<dim3(SCd, Bq * Hh), 128, 0, stream>>>(sc, SCd, SCd, scale, causal);
        // ctx[b,h] = P @ V_h          (K=300, kn layout)
        gemm(sc, vb, nullptr, t1, SCd, HD, SCd, SCd, Dd, Dd, /*bkn=*/1,
             Bq * Hh, Hh,
             (ll)Hh * SCd * SCd, (ll)SCd * SCd,
             (ll)SCd * Dd, HD, (ll)SCd * Dd, HD);
        gemm(t1, ow, ob, outbuf, T, Dd, Dd, Dd, Dd, Dd, 0);
    };

    // dense top-2 MoE: accumulates weighted expert outputs into t2
    auto moe = [&](const float* hx, const float* gw, const float* gb,
                   const float* w1, const float* b1, const float* w2,
                   const float* b2, const float* w3, const float* b3) {
        gate_kernel<<<T, 128, 0, stream>>>(hx, gw, gb, cw, Dd, Ee);
        for (int e = 0; e < Ee; ++e) {
            gemm(hx, w1 + (ll)e * Dd * Ff, b1 + (ll)e * Ff, h1, T, Ff, Dd, Dd, Ff, Ff, 1);
            gemm(hx, w2 + (ll)e * Dd * Ff, b2 + (ll)e * Ff, h2, T, Ff, Dd, Dd, Ff, Ff, 1);
            silu_mul_kernel<<<blocks((ll)T * Ff), 256, 0, stream>>>(h1, h2, (ll)T * Ff);
            gemm(h1, w3 + (ll)e * Ff * Dd, b3 + (ll)e * Dd, t1, T, Dd, Ff, Ff, Dd, Dd, 1);
            combine_kernel<<<blocks((ll)T * Dd), 256, 0, stream>>>(
                t2, t1, cw, e, Ee, Dd, (ll)T * Dd, e == 0 ? 1 : 0);
        }
    };

    // ---- input embeddings ----
    pack_vfc_kernel<<<blocks((ll)T * 776), 256, 0, stream>>>(
        fsem, fscene, fmotion, femotion, h1, (ll)T * 776);
    gemm(h1, w_vis, b_vis, hEnc, T, Dd, 776, 776, 776, Dd, 0);
    add_pos_kernel<<<blocks((ll)T * Dd), 256, 0, stream>>>(hEnc, pos_v, SVd, Dd, (ll)T * Dd);

    pack_chord_kernel<<<blocks((ll)T * 513), 256, 0, stream>>>(
        x_root, x_attr, fkey, emb_root, emb_attr, h2, SCd, (ll)T * 513);
    gemm(h2, w_chord, b_chord, hDec, T, Dd, 513, 513, 513, Dd, 0);
    add_pos_kernel<<<blocks((ll)T * Dd), 256, 0, stream>>>(hDec, pos_c, SCd, Dd, (ll)T * Dd);

    // ---- encoder ----
    for (int i = 0; i < Ln; ++i) {
        attention(hEnc, hEnc,
                  enc_qkv_w + (ll)i * 3 * Dd * Dd, enc_qkv_b + (ll)i * 3 * Dd,
                  enc_ow + (ll)i * Dd * Dd, enc_ob + (ll)i * Dd,
                  /*causal=*/0, t2);
        resln_kernel<<<T, 128, 0, stream>>>(hEnc, t2,
            enc_ln1_g + (ll)i * Dd, enc_ln1_b + (ll)i * Dd, hEnc, Dd);
        moe(hEnc, enc_gw + (ll)i * Ee * Dd, enc_gb + (ll)i * Ee,
            enc_w1 + (ll)i * Ee * Dd * Ff, enc_b1 + (ll)i * Ee * Ff,
            enc_w2 + (ll)i * Ee * Dd * Ff, enc_b2 + (ll)i * Ee * Ff,
            enc_w3 + (ll)i * Ee * Ff * Dd, enc_b3 + (ll)i * Ee * Dd);
        resln_kernel<<<T, 128, 0, stream>>>(hEnc, t2,
            enc_ln2_g + (ll)i * Dd, enc_ln2_b + (ll)i * Dd, hEnc, Dd);
    }
    resln_kernel<<<T, 128, 0, stream>>>(hEnc, nullptr, enc_fg, enc_fb, mem, Dd);

    // ---- decoder ----
    for (int i = 0; i < Ln; ++i) {
        attention(hDec, hDec,
                  dec_sqkv_w + (ll)i * 3 * Dd * Dd, dec_sqkv_b + (ll)i * 3 * Dd,
                  dec_sow + (ll)i * Dd * Dd, dec_sob + (ll)i * Dd,
                  /*causal=*/1, t2);
        resln_kernel<<<T, 128, 0, stream>>>(hDec, t2,
            dec_ln1_g + (ll)i * Dd, dec_ln1_b + (ll)i * Dd, hDec, Dd);
        attention(hDec, mem,
                  dec_cqkv_w + (ll)i * 3 * Dd * Dd, dec_cqkv_b + (ll)i * 3 * Dd,
                  dec_cow + (ll)i * Dd * Dd, dec_cob + (ll)i * Dd,
                  /*causal=*/0, t2);
        resln_kernel<<<T, 128, 0, stream>>>(hDec, t2,
            dec_ln2_g + (ll)i * Dd, dec_ln2_b + (ll)i * Dd, hDec, Dd);
        moe(hDec, dec_gw + (ll)i * Ee * Dd, dec_gb + (ll)i * Ee,
            dec_w1 + (ll)i * Ee * Dd * Ff, dec_b1 + (ll)i * Ee * Ff,
            dec_w2 + (ll)i * Ee * Dd * Ff, dec_b2 + (ll)i * Ee * Ff,
            dec_w3 + (ll)i * Ee * Ff * Dd, dec_b3 + (ll)i * Ee * Dd);
        resln_kernel<<<T, 128, 0, stream>>>(hDec, t2,
            dec_ln3_g + (ll)i * Dd, dec_ln3_b + (ll)i * Dd, hDec, Dd);
    }
    resln_kernel<<<T, 128, 0, stream>>>(hDec, nullptr, dec_fg, dec_fb, hDec, Dd);

    // ---- output projection ----
    gemm(hDec, wout, bout, out, T, CHORD, Dd, Dd, Dd, CHORD, 0);
}
